// BasicAttentionScratch_19232863551916
// MI455X (gfx1250) — matrix operand order repaired and checked by reading, not compiled, not run on hardware
//
#include <hip/hip_runtime.h>
#include <hip/hip_bf16.h>
#include <math.h>

#define SEQ 4096
#define BB  32
#define HH  512

typedef __attribute__((ext_vector_type(16))) __bf16    v16bf;
typedef __attribute__((ext_vector_type(8)))  float     v8f;
typedef __attribute__((ext_vector_type(4)))  float     v4f;
typedef __attribute__((ext_vector_type(8)))  unsigned  v8u;
typedef __attribute__((ext_vector_type(4)))  unsigned  v4u;

union ABits { v8u u; v16bf h; };

__device__ __forceinline__ unsigned bfp(float a, float b) {
    union { float f; unsigned u; } x, y;
    x.f = a; y.f = b;
    unsigned lo = (x.u + 0x7FFFu + ((x.u >> 16) & 1u)) >> 16;  // RNE f32->bf16
    unsigned hi = (y.u + 0x7FFFu + ((y.u >> 16) & 1u)) >> 16;
    return lo | (hi << 16);
}

__device__ __forceinline__ float fast_tanh(float x) {
#if defined(__AMDGCN__) && __has_builtin(__builtin_amdgcn_tanhf)
    return __builtin_amdgcn_tanhf(x);   // v_tanh_f32 on gfx1250
#else
    return ::tanhf(x);
#endif
}

// CDNA5 async global->LDS copy (ASYNCcnt path; no VGPR staging)
__device__ __forceinline__ void async_copy_b128(unsigned lds_off, const void* gaddr) {
    asm volatile("global_load_async_to_lds_b128 %0, %1, off"
                 :: "v"(lds_off), "v"((unsigned long long)(uintptr_t)gaddr)
                 : "memory");
}
__device__ __forceinline__ void wait_async0() {
#if defined(__AMDGCN__) && __has_builtin(__builtin_amdgcn_s_wait_asynccnt)
    __builtin_amdgcn_s_wait_asynccnt(0);
#else
    asm volatile("s_wait_asynccnt 0x0" ::: "memory");
#endif
}

// ---------------------------------------------------------------------------
// Kernel 1: h_proj[b][o] = sum_h hidden[b][h] * attn_W[o][h] + attn_b[o]
// ---------------------------------------------------------------------------
__global__ __launch_bounds__(256) void hproj_k(const float* __restrict__ hidden,
                                               const float* __restrict__ attn_W,
                                               const float* __restrict__ attn_b,
                                               float* __restrict__ hproj) {
    __shared__ float hb[HH];
    int b = blockIdx.x, tid = threadIdx.x;
    for (int i = tid; i < HH; i += 256) hb[i] = hidden[b * HH + i];
    __syncthreads();
    for (int o = tid; o < HH; o += 256) {
        const float* wrow = attn_W + (size_t)o * (2 * HH);   // Wh row o
        float acc = attn_b[o];
        #pragma unroll 4
        for (int k = 0; k < HH; ++k) acc = fmaf(hb[k], wrow[k], acc);
        hproj[b * HH + o] = acc;
    }
}

// ---------------------------------------------------------------------------
// Kernel 2: pack We (= attn_W[:, H:]) into bf16 WMMA B-fragment tile layout.
// Tile t (n = 16t..16t+15), chunk c (k = 32c..32c+31):
//   lane l holds column n = 16t + (l%16), k offset kh = (l>>4)*8;
//   frag dword j (0..7) packs (k, k+1) with k = 32c + kh + 16*(j>>2) + 2*(j&3).
// Flat dword index inside tile: ((c*2 + (j>>2))*32 + l)*4 + (j&3)
// ---------------------------------------------------------------------------
__global__ __launch_bounds__(256) void pack_we(const float* __restrict__ attn_W,
                                               unsigned* __restrict__ bpack) {
    int x = blockIdx.x * 256 + threadIdx.x;          // 131072 dwords total
    int j = x & 7, l = (x >> 3) & 31, c = (x >> 8) & 15, t = x >> 12;
    int n = t * 16 + (l & 15);
    int k = c * 32 + ((l >> 4) * 8) + (j >> 2) * 16 + 2 * (j & 3);
    const float* wr = attn_W + (size_t)n * (2 * HH) + HH + k;   // We[n][k], We[n][k+1]
    unsigned v = bfp(wr[0], wr[1]);
    int dw = t * 4096 + ((c * 2 + (j >> 2)) * 32 + l) * 4 + (j & 3);
    bpack[dw] = v;
}

// ---------------------------------------------------------------------------
// Kernel 3: fused  E-tile GEMM (bf16 WMMA) + tanh + dot(v)  -> scores[s*32+b]
// One wave32 per 16-row tile of the (S*B, H) row space; 8 waves / block.
// B fragments double-buffered through LDS via async global->LDS copies.
// ---------------------------------------------------------------------------
__global__ __launch_bounds__(256) void fused_scores(const float*    __restrict__ enc,
                                                    const float*    __restrict__ hproj,
                                                    const unsigned* __restrict__ bpack,
                                                    const float*    __restrict__ vW,
                                                    float*          __restrict__ scores) {
    __shared__ unsigned ldsB[2][4096];                // 2 x 16 KB B tiles
    const int tid  = threadIdx.x;
    const int wave = tid >> 5;
    const int lane = tid & 31;
    const int m    = lane & 15;                       // A row within tile
    const int hf   = lane >> 4;                       // lane half (k-group select)
    const long tileRow = ((long)blockIdx.x * 8 + wave) * 16;

    // per-thread staging slice: 64 B = 4 x b128, identical LDS/global layouts
    const char* gBbytes = (const char*)bpack;
    unsigned lds0 = (unsigned)(uintptr_t)(void*)&ldsB[0][tid * 16];
    unsigned lds1 = (unsigned)(uintptr_t)(void*)&ldsB[1][tid * 16];

    // ---- kick off async copy of B tile 0 while we convert the A tile
    #pragma unroll
    for (int q = 0; q < 4; ++q)
        async_copy_b128(lds0 + q * 16, gBbytes + (size_t)tid * 64 + q * 16);

    // ---- Load & convert the 16x512 f32 A tile into 16 bf16 A fragments (regs)
    v8u a[16];
    const float* arow = enc + (size_t)(tileRow + m) * HH + hf * 8;
    #pragma unroll
    for (int c = 0; c < 16; ++c) {
        const float* p = arow + c * 32;
        v4f x0 = *(const v4f*)(p);          // k = kh+0..3
        v4f x1 = *(const v4f*)(p + 4);      // k = kh+4..7
        v4f x2 = *(const v4f*)(p + 16);     // k = kh+16..19
        v4f x3 = *(const v4f*)(p + 20);     // k = kh+20..23
        v8u u;
        u[0] = bfp(x0[0], x0[1]); u[1] = bfp(x0[2], x0[3]);
        u[2] = bfp(x1[0], x1[1]); u[3] = bfp(x1[2], x1[3]);
        u[4] = bfp(x2[0], x2[1]); u[5] = bfp(x2[2], x2[3]);
        u[6] = bfp(x3[0], x3[1]); u[7] = bfp(x3[2], x3[3]);
        a[c] = u;
    }

    wait_async0();
    __syncthreads();

    const int b0 = (int)(tileRow & 16);               // row_base % 32
    const int n  = lane & 15;
    float sacc[8] = {0.f, 0.f, 0.f, 0.f, 0.f, 0.f, 0.f, 0.f};

    for (int t = 0; t < 32; ++t) {
        const int cur  = t & 1;
        const bool more = (t + 1) < 32;

        // issue async copy for next B tile (overlaps the 16 WMMAs below)
        if (more) {
            unsigned ldsn = cur ? lds0 : lds1;
            const char* src = gBbytes + (size_t)(t + 1) * 16384 + (size_t)tid * 64;
            #pragma unroll
            for (int q = 0; q < 4; ++q)
                async_copy_b128(ldsn + q * 16, src + q * 16);
        }

        // ---- 16 x v_wmma_f32_16x16x32_bf16 : acc = A(16x512) * We^T(:,16t..)
        v8f acc = {};
        #pragma unroll
        for (int c = 0; c < 16; ++c) {
            v4u blo = *(const v4u*)&ldsB[cur][((c * 2 + 0) * 32 + lane) * 4];
            v4u bhi = *(const v4u*)&ldsB[cur][((c * 2 + 1) * 32 + lane) * 4];
            ABits av, bv;
            av.u = a[c];
            bv.u = (v8u){blo[0], blo[1], blo[2], blo[3], bhi[0], bhi[1], bhi[2], bhi[3]};
            acc = __builtin_amdgcn_wmma_f32_16x16x32_bf16(
                      false, av.h, false, bv.h, (short)0, acc, false, false);
        }

        // ---- fused epilogue: tanh(acc + h_proj) * v, accumulate per-lane
        const int o  = t * 16 + n;
        const float vv = vW[o];
        #pragma unroll
        for (int i = 0; i < 8; ++i) {
            float hp = hproj[(b0 + i + 8 * hf) * HH + o];
            sacc[i] = fmaf(fast_tanh(acc[i] + hp), vv, sacc[i]);
        }

        if (more) wait_async0();
        __syncthreads();
    }

    // ---- cross-lane reduce over each 16-lane half (N dimension)
    #pragma unroll
    for (int i = 0; i < 8; ++i) {
        float x = sacc[i];
        x += __shfl_xor(x, 1, 32);
        x += __shfl_xor(x, 2, 32);
        x += __shfl_xor(x, 4, 32);
        x += __shfl_xor(x, 8, 32);
        sacc[i] = x;
    }
    if ((lane & 15) == 0) {
        #pragma unroll
        for (int i = 0; i < 8; ++i)
            scores[tileRow + i + 8 * hf] = sacc[i];   // flat index r = s*32 + b
    }
}

// ---------------------------------------------------------------------------
// Kernel 4: softmax over S for each b:  attn[b][s] = softmax_s(scores[s][b])
// ---------------------------------------------------------------------------
__global__ __launch_bounds__(256) void softmax_k(const float* __restrict__ scores,
                                                 float* __restrict__ attn) {
    __shared__ float red[256];
    int b = blockIdx.x, tid = threadIdx.x;
    float mx = -INFINITY;
    for (int s = tid; s < SEQ; s += 256) mx = fmaxf(mx, scores[s * BB + b]);
    red[tid] = mx; __syncthreads();
    for (int st = 128; st > 0; st >>= 1) {
        if (tid < st) red[tid] = fmaxf(red[tid], red[tid + st]);
        __syncthreads();
    }
    mx = red[0]; __syncthreads();
    float sum = 0.f;
    for (int s = tid; s < SEQ; s += 256) sum += __expf(scores[s * BB + b] - mx);
    red[tid] = sum; __syncthreads();
    for (int st = 128; st > 0; st >>= 1) {
        if (tid < st) red[tid] += red[tid + st];
        __syncthreads();
    }
    float inv = 1.f / red[0];
    for (int s = tid; s < SEQ; s += 256)
        attn[(size_t)b * SEQ + s] = __expf(scores[s * BB + b] - mx) * inv;
}

// ---------------------------------------------------------------------------
// Kernel 5: partial context over 64 s-chunks (streams enc a second time)
// ---------------------------------------------------------------------------
__global__ __launch_bounds__(256) void ctx_partial(const float* __restrict__ enc,
                                                   const float* __restrict__ attn,
                                                   float* __restrict__ partial) {
    int h  = blockIdx.x * 256 + threadIdx.x;          // gridDim.x = 2
    int b  = blockIdx.y;                              // 32
    int ch = blockIdx.z;                              // 64 chunks x 64 s
    const float* base = enc + ((size_t)(ch * 64) * BB + b) * HH + h;
    const float* w    = attn + (size_t)b * SEQ + ch * 64;
    float acc = 0.f;
    #pragma unroll 4
    for (int si = 0; si < 64; ++si)
        acc = fmaf(w[si], base[(size_t)si * BB * HH], acc);
    partial[((size_t)ch * BB + b) * HH + h] = acc;
}

__global__ __launch_bounds__(256) void ctx_reduce(const float* __restrict__ partial,
                                                  float* __restrict__ ctx) {
    int idx = blockIdx.x * 256 + threadIdx.x;         // 16384 outputs
    float acc = 0.f;
    #pragma unroll 4
    for (int ch = 0; ch < 64; ++ch) acc += partial[(size_t)ch * (BB * HH) + idx];
    ctx[idx] = acc;
}

// ---------------------------------------------------------------------------
extern "C" void kernel_launch(void* const* d_in, const int* in_sizes, int n_in,
                              void* d_out, int out_size, void* d_ws, size_t ws_size,
                              hipStream_t stream) {
    (void)in_sizes; (void)n_in; (void)out_size; (void)ws_size;
    const float* hidden = (const float*)d_in[0];   // (32, 512)
    const float* enc    = (const float*)d_in[1];   // (4096, 32, 512)
    const float* attn_W = (const float*)d_in[2];   // (512, 1024)
    const float* attn_b = (const float*)d_in[3];   // (512,)
    const float* vW     = (const float*)d_in[4];   // (1, 512)

    float* ctx_out  = (float*)d_out;               // context: 32*512
    float* attn_out = (float*)d_out + BB * HH;     // attn_weights: 32*4096

    char*     ws      = (char*)d_ws;
    float*    hproj   = (float*)(ws);                  //  64 KB
    unsigned* bpack   = (unsigned*)(ws + 65536);       // 512 KB
    float*    scores  = (float*)(ws + 589824);         // 512 KB
    float*    partial = (float*)(ws + 1114112);        //   4 MB

    hproj_k     <<<32,   256, 0, stream>>>(hidden, attn_W, attn_b, hproj);
    pack_we     <<<512,  256, 0, stream>>>(attn_W, bpack);
    fused_scores<<<1024, 256, 0, stream>>>(enc, hproj, bpack, vW, scores);
    softmax_k   <<<32,   256, 0, stream>>>(scores, attn_out);
    ctx_partial <<<dim3(2, 32, 64), 256, 0, stream>>>(enc, attn_out, partial);
    ctx_reduce  <<<64,   256, 0, stream>>>(partial, ctx_out);
}
